// CasualAttention_20461224198268
// MI455X (gfx1250) — compile-verified, matrix operand order checked by reading, not run on hardware
//
#include <hip/hip_runtime.h>

typedef _Float16 v8h  __attribute__((ext_vector_type(8)));
typedef _Float16 v16h __attribute__((ext_vector_type(16)));
typedef float    v8f  __attribute__((ext_vector_type(8)));
typedef float    v4f  __attribute__((ext_vector_type(4)));

#define NEGF (-1.0e9f)
#define S_LEN 1024
#define D_MODEL 1024
#define NHEAD 16
#define DK 64
#define BATCH 8
#define BH (BATCH * NHEAD)    // 128
#define MROWS (BATCH * S_LEN) // 8192

// ---- CDNA5 WMMA helpers -----------------------------------------------------

static __device__ __forceinline__ v8f wmma32(v16h a, v16h b, v8f c) {
  // v_wmma_f32_16x16x32_f16  D = A(16x32) * B(32x16) + C
  return __builtin_amdgcn_wmma_f32_16x16x32_f16(false, a, false, b, (short)0, c,
                                                false, false);
}

// 16-bit fragment load (A or B role): per lane, halves 0-7 = K in
// [k0+g*8, ...+8), halves 8-15 = K in [k0+16+g*8, ...+8). Row pointer is the
// lane's M (A) / N (B) row; both chunks are 16B-aligned b128 loads.
static __device__ __forceinline__ v16h frag_ld(const _Float16* row, int k0, int g) {
  v8h lo = *(const v8h*)(row + k0 + g * 8);
  v8h hi = *(const v8h*)(row + k0 + 16 + g * 8);
  return __builtin_shufflevector(lo, hi, 0, 1, 2, 3, 4, 5, 6, 7, 8, 9, 10, 11,
                                 12, 13, 14, 15);
}

// ---- Kernel 1: fp32 -> f16 conversion + mask bias ---------------------------

__global__ void __launch_bounds__(256) k_convert(
    const float* __restrict__ X, const float* __restrict__ Wq,
    const float* __restrict__ Wk, const float* __restrict__ Wv,
    const float* __restrict__ Wo, const int* __restrict__ smask,
    _Float16* __restrict__ Xh, _Float16* __restrict__ Wqh,
    _Float16* __restrict__ Wkh, _Float16* __restrict__ Wvh,
    _Float16* __restrict__ Woh, float* __restrict__ maskadd) {
  const size_t idx = (size_t)blockIdx.x * 256 + threadIdx.x;
  const size_t stride = (size_t)gridDim.x * 256;
  const int y = blockIdx.y;
  if (y == 0) {
    for (size_t i = idx; i < (size_t)MROWS * D_MODEL; i += stride)
      Xh[i] = (_Float16)X[i];
  } else if (y == 1) {
    for (size_t i = idx; i < (size_t)D_MODEL * D_MODEL; i += stride)
      Wqh[i] = (_Float16)Wq[i];
  } else if (y == 2) {
    for (size_t i = idx; i < (size_t)D_MODEL * D_MODEL; i += stride)
      Wkh[i] = (_Float16)Wk[i];
  } else if (y == 3) {
    for (size_t i = idx; i < (size_t)D_MODEL * D_MODEL; i += stride)
      Wvh[i] = (_Float16)Wv[i];
  } else if (y == 4) {
    for (size_t i = idx; i < (size_t)D_MODEL * D_MODEL; i += stride)
      Woh[i] = (_Float16)Wo[i];
  } else {
    for (size_t i = idx; i < (size_t)BATCH * S_LEN; i += stride)
      maskadd[i] = smask[i] ? 0.0f : NEGF;
  }
}

// ---- Kernel 2: fused QKV projection (Y = X @ W^T + b) -----------------------
// grid.y = z in {0:Q, 1:K, 2:V}. Each wave computes a 16(M) x 64(N) tile.
// Ping-pong double-buffered pipeline (no register-rotation copies): while the
// WMMAs consume buffer A, buffer B's loads are in flight, and vice versa.
// Q pre-scaled by 1/sqrt(dk). Q,K stored [b,h,s,dk]; V transposed [b,h,dk,s]
// so the attention PV B-fragment is key-contiguous.

__global__ void __launch_bounds__(256) k_qkv(
    const _Float16* __restrict__ Xh, const _Float16* __restrict__ Wqh,
    const _Float16* __restrict__ Wkh, const _Float16* __restrict__ Wvh,
    const float* __restrict__ bq, const float* __restrict__ bk,
    const float* __restrict__ bv, _Float16* __restrict__ Qh,
    _Float16* __restrict__ Kh, _Float16* __restrict__ Vt) {
  const int lane = threadIdx.x & 31;
  const int wave = threadIdx.x >> 5;
  const int g = lane >> 4;
  const int li = lane & 15;
  const int z = blockIdx.y;
  const int unit = blockIdx.x * 8 + wave;   // 0..8191
  const int m0 = (unit >> 4) * 16;          // row tile (token rows)
  const int n0 = (unit & 15) * 64;          // col tile (one head's 64 cols)

  const _Float16* W = (z == 0) ? Wqh : (z == 1) ? Wkh : Wvh;
  const float* bias = (z == 0) ? bq : (z == 1) ? bk : bv;

  const _Float16* arow = Xh + (size_t)(m0 + li) * D_MODEL;
  const _Float16* brow[4];
#pragma unroll
  for (int j = 0; j < 4; ++j)
    brow[j] = W + (size_t)(n0 + j * 16 + li) * D_MODEL;

  v8f acc[4] = {v8f{}, v8f{}, v8f{}, v8f{}};

  // ping-pong pipeline: set A for even k-steps, set B for odd k-steps
  v16h aA = frag_ld(arow, 0, g);
  v16h bA[4];
#pragma unroll
  for (int j = 0; j < 4; ++j) bA[j] = frag_ld(brow[j], 0, g);

  for (int k0 = 0; k0 < D_MODEL; k0 += 64) {   // 16 iterations of 2 k-steps
    const int k1 = k0 + 32;                    // always < D_MODEL
    v16h aB = frag_ld(arow, k1, g);
    v16h bB[4];
#pragma unroll
    for (int j = 0; j < 4; ++j) bB[j] = frag_ld(brow[j], k1, g);
#pragma unroll
    for (int j = 0; j < 4; ++j) acc[j] = wmma32(aA, bA[j], acc[j]);

    const int k2 = (k0 + 64 < D_MODEL) ? (k0 + 64) : 0;  // clamp: branch-free
    aA = frag_ld(arow, k2, g);
#pragma unroll
    for (int j = 0; j < 4; ++j) bA[j] = frag_ld(brow[j], k2, g);
#pragma unroll
    for (int j = 0; j < 4; ++j) acc[j] = wmma32(aB, bB[j], acc[j]);
  }

  const int b = m0 >> 10;        // batch
  const int s0 = m0 & 1023;      // sequence offset
  const float scale = (z == 0) ? 0.125f : 1.0f;  // 1/sqrt(64)
#pragma unroll
  for (int j = 0; j < 4; ++j) {
    const int n = n0 + j * 16 + li;
    const float bb = bias[n];
    const int h = n >> 6, d = n & 63;
#pragma unroll
    for (int r = 0; r < 8; ++r) {
      const int s = s0 + g * 8 + r;
      const float v = (acc[j][r] + bb) * scale;
      if (z == 2)
        Vt[(((size_t)b * NHEAD + h) * DK + d) * S_LEN + s] = (_Float16)v;
      else if (z == 0)
        Qh[(((size_t)b * NHEAD + h) * S_LEN + s) * DK + d] = (_Float16)v;
      else
        Kh[(((size_t)b * NHEAD + h) * S_LEN + s) * DK + d] = (_Float16)v;
    }
  }
}

// ---- Kernel 3: per-(b,h) column mean of V (for fully-masked rows) -----------

__global__ void __launch_bounds__(64) k_vmean(const _Float16* __restrict__ Vt,
                                              float* __restrict__ Vmean) {
  const int bh = blockIdx.x;
  const int d = threadIdx.x;
  const v8h* p = (const v8h*)(Vt + ((size_t)bh * DK + d) * S_LEN);
  float s = 0.f;
  for (int i = 0; i < S_LEN / 8; ++i) {
    v8h x = p[i];
#pragma unroll
    for (int t = 0; t < 8; ++t) s += (float)x[t];
  }
  Vmean[bh * DK + d] = s * (1.0f / S_LEN);
}

// ---- Kernel 4: causal flash attention, transposed-score trick ---------------
// One wave owns 16 queries. Scores computed as S^T = K * Q^T so lane = query,
// VGPRs = key rows; softmax stats live per lane; P^T C-tiles repack per-lane
// into the A-fragment of P for O += P*V. V-fragment loads are issued before
// the score WMMAs and next-block K-fragments are issued before the softmax
// VALU work, so memory latency hides behind compute.

__global__ void __launch_bounds__(128) k_attn(
    const _Float16* __restrict__ Qh, const _Float16* __restrict__ Kh,
    const _Float16* __restrict__ Vt, const float* __restrict__ maskadd,
    const float* __restrict__ Vmean, _Float16* __restrict__ Ch) {
  const int lane = threadIdx.x & 31;
  const int wave = threadIdx.x >> 5;
  const int g = lane >> 4;
  const int li = lane & 15;
  const int bh = blockIdx.y;
  const int b = bh >> 4;
  const int h = bh & 15;
  const int q0 = (blockIdx.x * 4 + wave) * 16;

  const _Float16* Qb = Qh + (size_t)bh * S_LEN * DK;
  const _Float16* Kb = Kh + (size_t)bh * S_LEN * DK;
  const _Float16* Vb = Vt + (size_t)bh * DK * S_LEN;
  const float* mrow = maskadd + (size_t)b * S_LEN;

  // B-fragments of Q^T (lane = query col), loop invariant
  const _Float16* qrow = Qb + (size_t)(q0 + li) * DK;
  const v16h bQ0 = frag_ld(qrow, 0, g);
  const v16h bQ1 = frag_ld(qrow, 32, g);

  v8f O[4] = {v8f{}, v8f{}, v8f{}, v8f{}};
  float m_run = -3.0e38f, l_run = 0.0f;
  const int qmy = q0 + li;               // this lane's query index
  const int nb = (q0 + 15) / 32 + 1;     // causal: key blocks of 32

  // K fragments double-buffered across blocks
  auto ldK = [&](int kb_, v16h* dst) {
    const _Float16* r0 = Kb + (size_t)(kb_ + li) * DK;
    dst[0] = frag_ld(r0, 0, g);
    dst[1] = frag_ld(r0, 32, g);
    const _Float16* r1 = Kb + (size_t)(kb_ + 16 + li) * DK;
    dst[2] = frag_ld(r1, 0, g);
    dst[3] = frag_ld(r1, 32, g);
  };
  v16h kfr[4];
  ldK(0, kfr);

  for (int blk = 0; blk < nb; ++blk) {
    const int kb = blk * 32;

    // V fragments for this block: independent of scores, issue first
    v16h vf[4];
#pragma unroll
    for (int j = 0; j < 4; ++j)
      vf[j] = frag_ld(Vb + (size_t)(j * 16 + li) * S_LEN, kb, g);

    // mask values (also independent): two 32B rows
    float ma[16];
    {
      const v4f* mp0 = (const v4f*)(mrow + kb + g * 8);
      const v4f* mp1 = (const v4f*)(mrow + kb + 16 + g * 8);
      v4f a0 = mp0[0], a1 = mp0[1], c0 = mp1[0], c1 = mp1[1];
#pragma unroll
      for (int t = 0; t < 4; ++t) {
        ma[t] = a0[t]; ma[t + 4] = a1[t];
        ma[t + 8] = c0[t]; ma[t + 12] = c1[t];
      }
    }

    // scores^T: two 16(key) x 16(query) tiles
    v8f T0 = {}, T1 = {};
    T0 = wmma32(kfr[0], bQ0, T0);
    T0 = wmma32(kfr[1], bQ1, T0);
    T1 = wmma32(kfr[2], bQ0, T1);
    T1 = wmma32(kfr[3], bQ1, T1);

    // prefetch next block's K fragments; they fly during the softmax VALU
    const int kbn = (blk + 1 < nb) ? (kb + 32) : 0;  // clamp: branch-free
    ldK(kbn, kfr);

    // causal + key-padding mask, exact -1e9 like the reference
#pragma unroll
    for (int r = 0; r < 8; ++r) {
      const int key0 = kb + g * 8 + r;
      const int key1 = kb + 16 + g * 8 + r;
      T0[r] = (key0 > qmy || ma[r] != 0.0f) ? NEGF : T0[r];
      T1[r] = (key1 > qmy || ma[r + 8] != 0.0f) ? NEGF : T1[r];
    }

    // online softmax (per-lane = per-query; pair lanes combine via xor 16)
    float bm = -3.0e38f;
#pragma unroll
    for (int r = 0; r < 8; ++r) { bm = fmaxf(bm, T0[r]); bm = fmaxf(bm, T1[r]); }
    bm = fmaxf(bm, __shfl_xor(bm, 16, 32));
    const float mnew = fmaxf(m_run, bm);
    const float corr = __expf(m_run - mnew);
    float rsum = 0.0f;
#pragma unroll
    for (int r = 0; r < 8; ++r) {
      T0[r] = __expf(T0[r] - mnew); rsum += T0[r];
      T1[r] = __expf(T1[r] - mnew); rsum += T1[r];
    }
    rsum += __shfl_xor(rsum, 16, 32);
    l_run = l_run * corr + rsum;
    m_run = mnew;

    // rescale O accumulators: factor per output row (= query g*8+r)
#pragma unroll
    for (int r = 0; r < 8; ++r) {
      const float cr = __shfl(corr, g * 8 + r, 32);
      O[0][r] *= cr; O[1][r] *= cr; O[2][r] *= cr; O[3][r] *= cr;
    }

    // P^T C-tiles -> A-fragment of P (pure per-lane f32->f16 pack)
    v8h plo, phi;
#pragma unroll
    for (int r = 0; r < 8; ++r) { plo[r] = (_Float16)T0[r]; phi[r] = (_Float16)T1[r]; }
    const v16h pa = __builtin_shufflevector(plo, phi, 0, 1, 2, 3, 4, 5, 6, 7, 8,
                                            9, 10, 11, 12, 13, 14, 15);

    // O += P * V  (B-frag from transposed V: lane = d, K = keys)
#pragma unroll
    for (int j = 0; j < 4; ++j) O[j] = wmma32(pa, vf[j], O[j]);
  }

  // epilogue: normalize by row sum, write ctx as f16 [b, s, h*64+d]
#pragma unroll
  for (int r = 0; r < 8; ++r) {
    const float lr = __shfl(l_run, g * 8 + r, 32);
    const float mr = __shfl(m_run, g * 8 + r, 32);
    const int s = q0 + g * 8 + r;
    _Float16* outp = Ch + ((size_t)b * S_LEN + s) * D_MODEL + h * DK;
    if (mr < -9.0e8f) {
      // fully-masked row: reference softmax is uniform over all S keys
#pragma unroll
      for (int j = 0; j < 4; ++j)
        outp[j * 16 + li] = (_Float16)Vmean[bh * DK + j * 16 + li];
    } else {
      const float inv = 1.0f / lr;
#pragma unroll
      for (int j = 0; j < 4; ++j)
        outp[j * 16 + li] = (_Float16)(O[j][r] * inv);
    }
  }
}

// ---- Kernel 5: output projection + residual + LayerNorm ---------------------
// Block = 16 rows x 1024 cols (16 waves x 64 cols) so LN fuses in-register.
// Same ping-pong pipeline as k_qkv.

__global__ void __launch_bounds__(512) k_proj_ln(
    const _Float16* __restrict__ Ch, const _Float16* __restrict__ Woh,
    const float* __restrict__ bo, const float* __restrict__ X,
    const float* __restrict__ gamma, const float* __restrict__ beta,
    float* __restrict__ out) {
  __shared__ float red[16][16][2];
  __shared__ float stats[16][2];
  const int lane = threadIdx.x & 31;
  const int wave = threadIdx.x >> 5;   // 0..15
  const int g = lane >> 4;
  const int li = lane & 15;
  const int m0 = blockIdx.x * 16;
  const int n0 = wave * 64;

  const _Float16* arow = Ch + (size_t)(m0 + li) * D_MODEL;
  const _Float16* brow[4];
#pragma unroll
  for (int j = 0; j < 4; ++j)
    brow[j] = Woh + (size_t)(n0 + j * 16 + li) * D_MODEL;

  v8f acc[4] = {v8f{}, v8f{}, v8f{}, v8f{}};

  v16h aA = frag_ld(arow, 0, g);
  v16h bA[4];
#pragma unroll
  for (int j = 0; j < 4; ++j) bA[j] = frag_ld(brow[j], 0, g);

  for (int k0 = 0; k0 < D_MODEL; k0 += 64) {   // 16 iterations of 2 k-steps
    const int k1 = k0 + 32;
    v16h aB = frag_ld(arow, k1, g);
    v16h bB[4];
#pragma unroll
    for (int j = 0; j < 4; ++j) bB[j] = frag_ld(brow[j], k1, g);
#pragma unroll
    for (int j = 0; j < 4; ++j) acc[j] = wmma32(aA, bA[j], acc[j]);

    const int k2 = (k0 + 64 < D_MODEL) ? (k0 + 64) : 0;
    aA = frag_ld(arow, k2, g);
#pragma unroll
    for (int j = 0; j < 4; ++j) bA[j] = frag_ld(brow[j], k2, g);
#pragma unroll
    for (int j = 0; j < 4; ++j) acc[j] = wmma32(aB, bB[j], acc[j]);
  }

  // bias + residual, then per-row partial sums over this wave's 64 cols
  float s1[8] = {}, s2[8] = {};
#pragma unroll
  for (int j = 0; j < 4; ++j) {
    const int n = n0 + j * 16 + li;
    const float bb = bo[n];
#pragma unroll
    for (int r = 0; r < 8; ++r) {
      const int m = m0 + g * 8 + r;
      float x = acc[j][r] + bb + X[(size_t)m * D_MODEL + n];
      acc[j][r] = x;
      s1[r] += x;
      s2[r] += x * x;
    }
  }
#pragma unroll
  for (int mask = 1; mask <= 8; mask <<= 1) {
#pragma unroll
    for (int r = 0; r < 8; ++r) {
      s1[r] += __shfl_xor(s1[r], mask, 32);
      s2[r] += __shfl_xor(s2[r], mask, 32);
    }
  }
  if (li == 0) {
#pragma unroll
    for (int r = 0; r < 8; ++r) {
      red[wave][g * 8 + r][0] = s1[r];
      red[wave][g * 8 + r][1] = s2[r];
    }
  }
  __syncthreads();
  if (threadIdx.x < 16) {
    float a = 0.f, b2 = 0.f;
#pragma unroll
    for (int w = 0; w < 16; ++w) {
      a += red[w][threadIdx.x][0];
      b2 += red[w][threadIdx.x][1];
    }
    const float mu = a * (1.0f / D_MODEL);
    const float var = b2 * (1.0f / D_MODEL) - mu * mu;
    stats[threadIdx.x][0] = mu;
    stats[threadIdx.x][1] = rsqrtf(var + 1e-5f);
  }
  __syncthreads();
#pragma unroll
  for (int j = 0; j < 4; ++j) {
    const int n = n0 + j * 16 + li;
    const float ga = gamma[n], be = beta[n];
#pragma unroll
    for (int r = 0; r < 8; ++r) {
      const int row = g * 8 + r;
      const int m = m0 + row;
      out[(size_t)m * D_MODEL + n] =
          (acc[j][r] - stats[row][0]) * stats[row][1] * ga + be;
    }
  }
}

// ---- Host launch ------------------------------------------------------------

extern "C" void kernel_launch(void* const* d_in, const int* in_sizes, int n_in,
                              void* d_out, int out_size, void* d_ws,
                              size_t ws_size, hipStream_t stream) {
  const float* X = (const float*)d_in[0];
  const int* smask = (const int*)d_in[1];
  const float* Wq = (const float*)d_in[2];
  const float* bq = (const float*)d_in[3];
  const float* Wk = (const float*)d_in[4];
  const float* bk = (const float*)d_in[5];
  const float* Wv = (const float*)d_in[6];
  const float* bv = (const float*)d_in[7];
  const float* Wo = (const float*)d_in[8];
  const float* bo = (const float*)d_in[9];
  const float* gamma = (const float*)d_in[10];
  const float* beta = (const float*)d_in[11];
  float* out = (float*)d_out;

  char* w = (char*)d_ws;
  _Float16* Xh = (_Float16*)w;  w += (size_t)MROWS * D_MODEL * 2;
  _Float16* Wqh = (_Float16*)w; w += (size_t)D_MODEL * D_MODEL * 2;
  _Float16* Wkh = (_Float16*)w; w += (size_t)D_MODEL * D_MODEL * 2;
  _Float16* Wvh = (_Float16*)w; w += (size_t)D_MODEL * D_MODEL * 2;
  _Float16* Woh = (_Float16*)w; w += (size_t)D_MODEL * D_MODEL * 2;
  _Float16* Qh = (_Float16*)w;  w += (size_t)MROWS * D_MODEL * 2;
  _Float16* Kh = (_Float16*)w;  w += (size_t)MROWS * D_MODEL * 2;
  _Float16* Vt = (_Float16*)w;  w += (size_t)MROWS * D_MODEL * 2;
  _Float16* Ch = (_Float16*)w;  w += (size_t)MROWS * D_MODEL * 2;
  float* maskadd = (float*)w;   w += (size_t)BATCH * S_LEN * 4;
  float* Vmean = (float*)w;     w += (size_t)BH * DK * 4;

  k_convert<<<dim3(4096, 6), 256, 0, stream>>>(X, Wq, Wk, Wv, Wo, smask, Xh,
                                               Wqh, Wkh, Wvh, Woh, maskadd);
  k_qkv<<<dim3(1024, 3), 256, 0, stream>>>(Xh, Wqh, Wkh, Wvh, bq, bk, bv, Qh,
                                           Kh, Vt);
  k_vmean<<<dim3(BH), 64, 0, stream>>>(Vt, Vmean);
  k_attn<<<dim3(16, BH), 128, 0, stream>>>(Qh, Kh, Vt, maskadd, Vmean, Ch);
  k_proj_ln<<<dim3(512), 512, 0, stream>>>(Ch, Woh, bo, X, gamma, beta, out);
}
